// SFAM_73461120631425
// MI455X (gfx1250) — matrix operand order repaired and checked by reading, not compiled, not run on hardware
//
#include <hip/hip_runtime.h>
#include <hip/hip_bf16.h>

// ---------------------------------------------------------------------------
// Fused stereo cross-attention (SFAM) for MI455X / gfx1250.
// One block per (b, h) row-pair. Everything stays in LDS (320 KB):
//   LN -> bf16 WMMA projections (Q_l, Q_r, V_l^T, V_r^T) -> two-direction
//   row-attention with online softmax -> residual epilogue.
// All matmuls via v_wmma_f32_16x16x32_bf16. Wave-uniform values are forced
// into SGPRs via readfirstlane so per-wave branching stays scalar.
// ---------------------------------------------------------------------------

typedef __attribute__((ext_vector_type(16))) __bf16 v16bf;
typedef __attribute__((ext_vector_type(8)))  __bf16 v8bf;
typedef __attribute__((ext_vector_type(8)))  float  v8f;

#define BATCH 4
#define CCH   64
#define HH    180
#define WW    320
#define HWSZ  (HH * WW)            // 57600
#define NTOT  ((size_t)BATCH * CCH * HH * WW)   // 14745600
#define LDSF  20480                // elements per 40KB bf16 buffer

__device__ __forceinline__ v8f wmma_bf16(v16bf a, v16bf b, v8f c) {
  return __builtin_amdgcn_wmma_f32_16x16x32_bf16(false, a, false, b,
                                                 (short)0, c, false, false);
}

// A-matrix tile (16xK=32, bf16) from a row-major [rows][ld] bf16 array.
// Contiguous-k layout (shared with loadB): lane<16 holds row m0+lane, K = k0..k0+15 contiguous;
// lane>=16 holds K = k0+16..k0+31.
__device__ __forceinline__ v16bf loadA_bf16(const __bf16* src, int ld,
                                            int m0, int k0, int lane) {
  int row  = m0 + (lane & 15);
  int koff = k0 + ((lane & 16) ? 16 : 0);
  const __bf16* p = src + row * ld + koff;
  v8bf lo = *(const v8bf*)p;
  v8bf hi = *(const v8bf*)(p + 8);
  v16bf a;
#pragma unroll
  for (int i = 0; i < 8; ++i) { a[i] = lo[i]; a[i + 8] = hi[i]; }
  return a;
}

// B-matrix tile (K=32 x 16, bf16) where B[k][n] = src[n][k], src row-major ld.
// Contiguous-k layout (shared with loadA): lane<16 holds column n0+lane with K = k0..k0+15 contiguous;
// lane>=16 holds K = k0+16..k0+31.
__device__ __forceinline__ v16bf loadB_bf16(const __bf16* src, int ld,
                                            int k0, int n0, int lane) {
  int n  = n0 + (lane & 15);
  int kk = k0 + ((lane & 16) ? 16 : 0);
  return *(const v16bf*)(src + n * ld + kk);
}

// Same B layout but sourced from a row-major f32 [64][64] weight matrix in
// global memory, converted to bf16 on the fly (B[k][n] = w[n][k]).
__device__ __forceinline__ v16bf loadB_wf32(const float* src,
                                            int k0, int n0, int lane) {
  int n  = n0 + (lane & 15);
  int kk = k0 + ((lane & 16) ? 16 : 0);
  const float* p = src + n * 64 + kk;
  v16bf b;
#pragma unroll
  for (int i = 0; i < 16; ++i) b[i] = (__bf16)p[i];
  return b;
}

// Reductions across the 16-lane half the calling lane lives in (wave32).
__device__ __forceinline__ float halfmax(float v) {
#pragma unroll
  for (int m = 1; m < 16; m <<= 1) v = fmaxf(v, __shfl_xor(v, m, 32));
  return v;
}
__device__ __forceinline__ float halfsum(float v) {
#pragma unroll
  for (int m = 1; m < 16; m <<= 1) v += __shfl_xor(v, m, 32);
  return v;
}

__global__ void __launch_bounds__(256, 1)
sfam_fused_kernel(const float* __restrict__ x_l, const float* __restrict__ x_r,
                  const float* __restrict__ ln_l_w, const float* __restrict__ ln_l_b,
                  const float* __restrict__ ln_r_w, const float* __restrict__ ln_r_b,
                  const float* __restrict__ wq_l, const float* __restrict__ bq_l,
                  const float* __restrict__ wq_r, const float* __restrict__ bq_r,
                  const float* __restrict__ wv_l, const float* __restrict__ bv_l,
                  const float* __restrict__ wv_r, const float* __restrict__ bv_r,
                  const float* __restrict__ beta, const float* __restrict__ gamma,
                  float* __restrict__ out) {
  extern __shared__ __align__(16) char smem_raw[];
  __bf16* sQl  = (__bf16*)smem_raw;      // Q_l  [W][C]
  __bf16* sQr  = sQl  + LDSF;            // Q_r  [W][C]
  __bf16* sVtl = sQr  + LDSF;            // V_l^T [C][W]
  __bf16* sVtr = sVtl + LDSF;            // V_r^T [C][W]
  __bf16* sXnl = sVtr + LDSF;            // LN(x_l) [W][C]  -> reused as F_r2l [C][W]
  __bf16* sXnr = sXnl + LDSF;            // LN(x_r) [W][C]  -> reused as F_l2r [C][W]
  __bf16* sXrl = sXnr + LDSF;            // raw x_l [W][C]  -> reused as P scratch
  __bf16* sXrr = sXrl + LDSF;            // raw x_r [W][C]

  const int tid   = threadIdx.x;
  const int lane  = tid & 31;
  // Wave index is uniform within a wave: force it into an SGPR so all
  // wave-level control flow / pointer selection stays scalar.
  const int waveU = __builtin_amdgcn_readfirstlane(tid >> 5);
  const int bh = blockIdx.x;
  const int b  = bh / HH;
  const int h  = bh % HH;

  // -------- Phase 0: load rows, LayerNorm, stage bf16 operands ------------
  for (int side = 0; side < 2; ++side) {
    const float* xg  = side ? x_r   : x_l;
    const float* lw  = side ? ln_r_w : ln_l_w;
    const float* lb  = side ? ln_r_b : ln_l_b;
    __bf16* sXraw = side ? sXrr : sXrl;
    __bf16* sXn   = side ? sXnr : sXnl;
    for (int w = tid; w < WW; w += 256) {
      const float* px = xg + (size_t)b * CCH * HWSZ + (size_t)h * WW + w;
      float sum = 0.f, sq = 0.f;
#pragma unroll 2
      for (int c8 = 0; c8 < 8; ++c8) {
        v8bf pk;
#pragma unroll
        for (int i = 0; i < 8; ++i) {
          float v = px[(size_t)(c8 * 8 + i) * HWSZ];
          sum += v; sq += v * v;
          pk[i] = (__bf16)v;
        }
        *(v8bf*)(sXraw + w * CCH + c8 * 8) = pk;
      }
      float mu   = sum * (1.0f / CCH);
      float var  = sq * (1.0f / CCH) - mu * mu;
      float rstd = rsqrtf(var + 1e-6f);
#pragma unroll 2
      for (int c8 = 0; c8 < 8; ++c8) {
        v8bf raw = *(const v8bf*)(sXraw + w * CCH + c8 * 8);
        v8bf pk;
#pragma unroll
        for (int i = 0; i < 8; ++i) {
          int c = c8 * 8 + i;
          float v  = (float)raw[i];
          float xn = (v - mu) * rstd * lw[c] + lb[c];
          pk[i] = (__bf16)xn;
        }
        *(v8bf*)(sXn + w * CCH + c8 * 8) = pk;
      }
    }
  }
  __syncthreads();

  // -------- Phase 1: Q/V projections (WMMA bf16, 64x64 weights) ----------
  // Each wave owns one matrix (mat = waveU & 3) and 10 of the 20 M-tiles;
  // the 8 weight B-tiles and biases are hoisted out of the M-loop.
  {
    const int mat = waveU & 3;
    const __bf16* Asrc = (mat == 0) ? sXnl : (mat == 1) ? sXnr
                       : (mat == 2) ? sXrl : sXrr;
    const float* Wg = (mat == 0) ? wq_l : (mat == 1) ? wq_r
                    : (mat == 2) ? wv_l : wv_r;
    const float* Bg = (mat == 0) ? bq_l : (mat == 1) ? bq_r
                    : (mat == 2) ? bv_l : bv_r;
    const bool isV = (mat >= 2);
    __bf16* dst = (mat == 0) ? sQl : (mat == 1) ? sQr
                : (mat == 2) ? sVtl : sVtr;

    v16bf wb0[4], wb1[4];
    float bias[4];
#pragma unroll
    for (int n = 0; n < 4; ++n) {
      wb0[n] = loadB_wf32(Wg, 0,  n * 16, lane);
      wb1[n] = loadB_wf32(Wg, 32, n * 16, lane);
      bias[n] = Bg[n * 16 + (lane & 15)];
    }

    const int mt0 = waveU >> 2;   // scalar: waves 0-3 even tiles, 4-7 odd
    for (int mt = mt0; mt < 20; mt += 2) {
      int w0 = mt * 16;
      v16bf a0 = loadA_bf16(Asrc, CCH, w0, 0, lane);
      v16bf a1 = loadA_bf16(Asrc, CCH, w0, 32, lane);
      int wbase = w0 + ((lane & 16) ? 8 : 0);
#pragma unroll
      for (int n = 0; n < 4; ++n) {
        v8f acc = {};
        acc = wmma_bf16(a0, wb0[n], acc);
        acc = wmma_bf16(a1, wb1[n], acc);
        int cc = n * 16 + (lane & 15);
        if (isV) {                       // V^T [C][W]: 8 contiguous bf16
          v8bf pk;
#pragma unroll
          for (int r = 0; r < 8; ++r) pk[r] = (__bf16)(acc[r] + bias[n]);
          *(v8bf*)(dst + cc * WW + wbase) = pk;
        } else {                         // Q [W][C]: strided b16 stores
#pragma unroll
          for (int r = 0; r < 8; ++r)
            dst[(wbase + r) * CCH + cc] = (__bf16)(acc[r] + bias[n]);
        }
      }
    }
  }
  __syncthreads();

  // -------- Phase 2: two-direction row attention (flash, two-pass) -------
  // 40 tasks = 2 directions x 20 query strips of 16; 5 tasks per wave.
  const float scale = 0.125f;  // C^-0.5
  __bf16* scr = sXrl + waveU * (16 * 32);  // per-wave P transpose buffer
  for (int task = waveU; task < 40; task += 8) {
    const int dir = task / 20;       // scalar; 0: q=Q_l (r2l), 1: q=Q_r (l2r)
    const int w0  = (task % 20) * 16;
    const __bf16* Qq = dir ? sQr  : sQl;
    const __bf16* Kk = dir ? sQl  : sQr;
    const __bf16* Vt = dir ? sVtl : sVtr;
    __bf16* F        = dir ? sXnr : sXnl;   // F [C][W]

    v16bf aq0 = loadA_bf16(Qq, CCH, w0, 0, lane);
    v16bf aq1 = loadA_bf16(Qq, CCH, w0, 32, lane);

    float mrow[8], lrow[8];
#pragma unroll
    for (int r = 0; r < 8; ++r) { mrow[r] = -3.0e38f; lrow[r] = 0.f; }

    // Pass 1: online row max + sum of exponentials.
    for (int j = 0; j < 20; ++j) {
      v16bf b0 = loadB_bf16(Kk, CCH, 0, j * 16, lane);
      v16bf b1 = loadB_bf16(Kk, CCH, 32, j * 16, lane);
      v8f s = {};
      s = wmma_bf16(aq0, b0, s);
      s = wmma_bf16(aq1, b1, s);
#pragma unroll
      for (int r = 0; r < 8; ++r) {
        float v  = s[r] * scale;
        float mx = halfmax(v);
        float nm = fmaxf(mrow[r], mx);
        float se = halfsum(__expf(v - nm));
        lrow[r]  = lrow[r] * __expf(mrow[r] - nm) + se;
        mrow[r]  = nm;
      }
    }

    // Pass 2: recompute scores, P = exp(s - m), accumulate O = P * V.
    v8f o0 = {}, o1 = {}, o2 = {}, o3 = {};
    for (int jc = 0; jc < 10; ++jc) {
#pragma unroll
      for (int jj = 0; jj < 2; ++jj) {
        int j = jc * 2 + jj;
        v16bf b0 = loadB_bf16(Kk, CCH, 0, j * 16, lane);
        v16bf b1 = loadB_bf16(Kk, CCH, 32, j * 16, lane);
        v8f s = {};
        s = wmma_bf16(aq0, b0, s);
        s = wmma_bf16(aq1, b1, s);
        int row0 = (lane & 16) ? 8 : 0;
        int col  = jj * 16 + (lane & 15);
#pragma unroll
        for (int r = 0; r < 8; ++r) {
          float p = __expf(s[r] * scale - mrow[r]);
          scr[(row0 + r) * 32 + col] = (__bf16)p;  // P in [16 rows][32 keys]
        }
      }
      v16bf ap = loadA_bf16(scr, 32, 0, 0, lane);
      o0 = wmma_bf16(ap, loadB_bf16(Vt, WW, jc * 32, 0,  lane), o0);
      o1 = wmma_bf16(ap, loadB_bf16(Vt, WW, jc * 32, 16, lane), o1);
      o2 = wmma_bf16(ap, loadB_bf16(Vt, WW, jc * 32, 32, lane), o2);
      o3 = wmma_bf16(ap, loadB_bf16(Vt, WW, jc * 32, 48, lane), o3);
    }

    // Normalize and stage F [C][W]: per lane 8 contiguous bf16 per n-tile.
    int wbase = w0 + ((lane & 16) ? 8 : 0);
    int cl    = lane & 15;
    float inv[8];
#pragma unroll
    for (int r = 0; r < 8; ++r) inv[r] = 1.0f / lrow[r];
    v8bf p0, p1, p2, p3;
#pragma unroll
    for (int r = 0; r < 8; ++r) {
      p0[r] = (__bf16)(o0[r] * inv[r]);
      p1[r] = (__bf16)(o1[r] * inv[r]);
      p2[r] = (__bf16)(o2[r] * inv[r]);
      p3[r] = (__bf16)(o3[r] * inv[r]);
    }
    *(v8bf*)(F + (0  + cl) * WW + wbase) = p0;
    *(v8bf*)(F + (16 + cl) * WW + wbase) = p1;
    *(v8bf*)(F + (32 + cl) * WW + wbase) = p2;
    *(v8bf*)(F + (48 + cl) * WW + wbase) = p3;
  }
  __syncthreads();

  // -------- Phase 3: residual epilogue, coalesced along W ----------------
  float* out_l = out;
  float* out_r = out + NTOT;
  for (int idx = tid; idx < CCH * WW; idx += 256) {
    int c = idx / WW;
    int w = idx - c * WW;
    size_t gi = ((size_t)(b * CCH + c) * HH + h) * WW + w;
    out_l[gi] = x_l[gi] + (float)sXnl[c * WW + w] * beta[c];
    out_r[gi] = x_r[gi] + (float)sXnr[c * WW + w] * gamma[c];
  }
}

extern "C" void kernel_launch(void* const* d_in, const int* in_sizes, int n_in,
                              void* d_out, int out_size, void* d_ws, size_t ws_size,
                              hipStream_t stream) {
  (void)in_sizes; (void)n_in; (void)out_size; (void)d_ws; (void)ws_size;
  const float* x_l    = (const float*)d_in[0];
  const float* x_r    = (const float*)d_in[1];
  const float* ln_l_w = (const float*)d_in[2];
  const float* ln_l_b = (const float*)d_in[3];
  const float* ln_r_w = (const float*)d_in[4];
  const float* ln_r_b = (const float*)d_in[5];
  const float* wq_l   = (const float*)d_in[6];
  const float* bq_l   = (const float*)d_in[7];
  const float* wq_r   = (const float*)d_in[8];
  const float* bq_r   = (const float*)d_in[9];
  const float* wv_l   = (const float*)d_in[10];
  const float* bv_l   = (const float*)d_in[11];
  const float* wv_r   = (const float*)d_in[12];
  const float* bv_r   = (const float*)d_in[13];
  const float* beta   = (const float*)d_in[14];
  const float* gamma  = (const float*)d_in[15];
  float* out = (float*)d_out;

  const int smem = 8 * LDSF * 2;  // 320 KB
  (void)hipFuncSetAttribute((const void*)sfam_fused_kernel,
                            hipFuncAttributeMaxDynamicSharedMemorySize, smem);
  sfam_fused_kernel<<<dim3(BATCH * HH), dim3(256), smem, stream>>>(
      x_l, x_r, ln_l_w, ln_l_b, ln_r_w, ln_r_b,
      wq_l, bq_l, wq_r, bq_r, wv_l, bv_l, wv_r, bv_r,
      beta, gamma, out);
}
